// SelfAttention_16269336117812
// MI455X (gfx1250) — matrix operand order repaired and checked by reading, not compiled, not run on hardware
//
#include <hip/hip_runtime.h>

typedef _Float16 v16h __attribute__((ext_vector_type(16)));
typedef float    v8f  __attribute__((ext_vector_type(8)));

constexpr int B_ = 2, S_ = 2048, D_ = 1024, H_ = 16, HD_ = 64;

__device__ __forceinline__ v8f wmma_f16(v16h a, v16h b, v8f c) {
  return __builtin_amdgcn_wmma_f32_16x16x32_f16(
      /*neg_a=*/false, a, /*neg_b=*/false, b,
      /*c_mod=*/(short)0, c, /*reuse_a=*/false, /*reuse_b=*/false);
}

// Wave-local LDS producer->consumer fence (LDS ops are in-order per wave,
// but cross-lane data needs the DS counter drained before the read).
__device__ __forceinline__ void wave_lds_sync() {
#if __has_builtin(__builtin_amdgcn_s_wait_dscnt)
  asm volatile("" ::: "memory");
  __builtin_amdgcn_s_wait_dscnt(0);
  asm volatile("" ::: "memory");
#else
  asm volatile("s_wait_dscnt 0" ::: "memory");
#endif
}

__device__ __forceinline__ void wait_async0() {
#if __has_builtin(__builtin_amdgcn_s_wait_asynccnt)
  asm volatile("" ::: "memory");
  __builtin_amdgcn_s_wait_asynccnt(0);
  asm volatile("" ::: "memory");
#else
  asm volatile("s_wait_asynccnt 0" ::: "memory");
#endif
}

// Async global->LDS copy, 16B per lane, tracked by ASYNCcnt (no VGPR data path).
__device__ __forceinline__ void async_copy_b128(unsigned lds_off, const _Float16* gaddr) {
  asm volatile("global_load_async_to_lds_b128 %0, %1, off"
               :: "v"(lds_off), "v"((unsigned long long)(uintptr_t)gaddr)
               : "memory");
}

// ---- WMMA fragment gathers from LDS (contiguous k = 16*hi + s order, wave32) ----

// A 16x32 (MxK): lane = (hi<<4)|m, VGPR v holds K = 16*hi + 2*v (+1)
__device__ __forceinline__ v16h load_A_frag(const _Float16* lds, int row0, int k0, int ldm) {
  const int lane = threadIdx.x & 31;
  const int m = lane & 15, hi = lane >> 4;
  const _Float16* base = lds + (row0 + m) * ldm + k0;
  v16h a;
#pragma unroll
  for (int v = 0; v < 8; ++v) {
    const int k = (hi << 4) + (v << 1);
    a[2 * v]     = base[k];
    a[2 * v + 1] = base[k + 1];
  }
  return a;
}

// B 32x16 (KxN) from row-major K-by-N LDS tile: lane = (hi<<4)|n,
// VGPR v holds K = 16*hi + 2*v (+1), column n.
__device__ __forceinline__ v16h load_B_frag(const _Float16* lds, int k0, int col0, int ldm) {
  const int lane = threadIdx.x & 31;
  const int n = lane & 15, hi = lane >> 4;
  const _Float16* base = lds + (k0 + (hi << 4)) * ldm + col0 + n;
  v16h b;
#pragma unroll
  for (int v = 0; v < 8; ++v) {
    b[2 * v]     = base[(2 * v) * ldm];
    b[2 * v + 1] = base[(2 * v + 1) * ldm];
  }
  return b;
}

// B 32x16 where the LDS tile is stored N-major (row = n, col = k), i.e. K^T
// for Q·K^T: element (k,n) = lds[(row0+n)*ldm + kcol0 + k] (contiguous in k).
__device__ __forceinline__ v16h load_B_fragT(const _Float16* lds, int row0, int kcol0, int ldm) {
  const int lane = threadIdx.x & 31;
  const int n = lane & 15, hi = lane >> 4;
  const _Float16* base = lds + (row0 + n) * ldm + kcol0 + (hi << 4);
  v16h b;
#pragma unroll
  for (int v = 0; v < 8; ++v) {
    b[2 * v]     = base[2 * v];
    b[2 * v + 1] = base[2 * v + 1];
  }
  return b;
}

// ----------------- pipelined tile fetch (global -> registers) -----------------
template <int MODE>
__device__ __forceinline__ void fetch_tiles(
    const float* __restrict__ A32, const _Float16* __restrict__ A16,
    const float* __restrict__ Bw, int row0, int col0, int k0, int N, int K,
    float4 (&fa)[4], uint4 (&ha)[2], float4 (&fb)[4]) {
  if (MODE == 0) {
#pragma unroll
    for (int i = 0; i < 4; ++i) {
      const int lin = i * 512 + (int)threadIdx.x * 4;
      const int r = lin >> 5, c = lin & 31;
      fa[i] = *(const float4*)(A32 + (size_t)(row0 + r) * K + k0 + c);
    }
  } else {
#pragma unroll
    for (int i = 0; i < 2; ++i) {
      const int lin = i * 1024 + (int)threadIdx.x * 8;
      const int r = lin >> 5, c = lin & 31;
      ha[i] = *(const uint4*)(A16 + (size_t)(row0 + r) * K + k0 + c);
    }
  }
#pragma unroll
  for (int i = 0; i < 4; ++i) {
    const int lin = i * 512 + (int)threadIdx.x * 4;
    const int r = lin >> 6, c = lin & 63;
    fb[i] = *(const float4*)(Bw + (size_t)(k0 + r) * N + col0 + c);
  }
}

// ------------------- generic WMMA GEMM: C = A*B + bias -------------------
// MODE 0: A = f32 (X), scatter f16 result into head-split Q/K/V buffers.
// MODE 1: A = f16 (attn out), write f32 result to outF32.
template <int MODE>
__global__ __launch_bounds__(128) void wmma_gemm_kernel(
    const float* __restrict__ A32, const _Float16* __restrict__ A16,
    const float* __restrict__ Bw, const float* __restrict__ bias,
    float* __restrict__ outF32,
    _Float16* __restrict__ Qh, _Float16* __restrict__ Kh, _Float16* __restrict__ Vh,
    int M, int N, int K) {
  (void)M;
  __shared__ _Float16 As[64][40];   // 64x32 tile, stride 40 (16B-aligned rows)
  __shared__ _Float16 Bs[32][72];   // 32x64 tile, stride 72

  const int row0 = blockIdx.y * 64;
  const int col0 = blockIdx.x * 64;
  const int wid  = threadIdx.x >> 5;
  const int lane = threadIdx.x & 31;
  const int wm = wid >> 1, wn = wid & 1;
  const int hi = lane >> 4, nl = lane & 15;

  v8f acc[2][2] = {};
  float4 fa[4]; uint4 ha[2]; float4 fb[4];

  // prologue: fetch first k-tile into registers
  fetch_tiles<MODE>(A32, A16, Bw, row0, col0, 0, N, K, fa, ha, fb);

  for (int k0 = 0; k0 < K; k0 += 32) {
    // commit staged registers to LDS (with f32 -> f16 conversion)
    if (MODE == 0) {
#pragma unroll
      for (int i = 0; i < 4; ++i) {
        const int lin = i * 512 + (int)threadIdx.x * 4;
        const int r = lin >> 5, c = lin & 31;
        As[r][c]     = (_Float16)fa[i].x; As[r][c + 1] = (_Float16)fa[i].y;
        As[r][c + 2] = (_Float16)fa[i].z; As[r][c + 3] = (_Float16)fa[i].w;
      }
    } else {
#pragma unroll
      for (int i = 0; i < 2; ++i) {
        const int lin = i * 1024 + (int)threadIdx.x * 8;
        const int r = lin >> 5, c = lin & 31;
        *(uint4*)&As[r][c] = ha[i];
      }
    }
#pragma unroll
    for (int i = 0; i < 4; ++i) {
      const int lin = i * 512 + (int)threadIdx.x * 4;
      const int r = lin >> 6, c = lin & 63;
      Bs[r][c]     = (_Float16)fb[i].x; Bs[r][c + 1] = (_Float16)fb[i].y;
      Bs[r][c + 2] = (_Float16)fb[i].z; Bs[r][c + 3] = (_Float16)fb[i].w;
    }
    __syncthreads();

    // overlap: fetch next k-tile from global while WMMAs run off LDS
    if (k0 + 32 < K)
      fetch_tiles<MODE>(A32, A16, Bw, row0, col0, k0 + 32, N, K, fa, ha, fb);

    const v16h a0 = load_A_frag(&As[0][0], wm * 32,      0, 40);
    const v16h a1 = load_A_frag(&As[0][0], wm * 32 + 16, 0, 40);
    const v16h b0 = load_B_frag(&Bs[0][0], 0, wn * 32,      72);
    const v16h b1 = load_B_frag(&Bs[0][0], 0, wn * 32 + 16, 72);
    acc[0][0] = wmma_f16(a0, b0, acc[0][0]);
    acc[0][1] = wmma_f16(a0, b1, acc[0][1]);
    acc[1][0] = wmma_f16(a1, b0, acc[1][0]);
    acc[1][1] = wmma_f16(a1, b1, acc[1][1]);
    __syncthreads();
  }

#pragma unroll
  for (int ti = 0; ti < 2; ++ti)
#pragma unroll
    for (int tj = 0; tj < 2; ++tj)
#pragma unroll
      for (int r = 0; r < 8; ++r) {
        const int grow = row0 + wm * 32 + ti * 16 + r + hi * 8;
        const int gcol = col0 + wn * 32 + tj * 16 + nl;
        const float val = acc[ti][tj][r] + bias[gcol];
        if (MODE == 0) {
          const int which = gcol >> 10;        // 0=Q 1=K 2=V
          const int d = gcol & 1023;
          const int hh = d >> 6, hd = d & 63;
          const int bb = grow >> 11, ss = grow & 2047;  // S_ = 2048
          _Float16* dst = (which == 0) ? Qh : ((which == 1) ? Kh : Vh);
          dst[(((size_t)bb * H_ + hh) * S_ + ss) * HD_ + hd] = (_Float16)val;
        } else {
          outF32[(size_t)grow * N + gcol] = val;
        }
      }
}

// ------------------------- attention (flash-style) -------------------------
// Block = 64 queries of one (b,h); 4 waves x 16 query rows each.
// Full softmax over all S keys; masked (k>q) scores forced to -0.0001 per ref.
// K/V staged with double-buffered GLOBAL_LOAD_ASYNC_TO_LDS (ASYNCcnt).
__global__ __launch_bounds__(128) void attention_kernel(
    const _Float16* __restrict__ Qh, const _Float16* __restrict__ Kh,
    const _Float16* __restrict__ Vh, _Float16* __restrict__ AO) {
  __shared__ _Float16 Ks[2][32][72];
  __shared__ _Float16 Vs[2][32][72];
  __shared__ _Float16 Ps[4][16][40];

  const int wid  = threadIdx.x >> 5;
  const int lane = threadIdx.x & 31;
  const int hi = lane >> 4, nl = lane & 15;

  const int qblk = blockIdx.x & 31;         // S/64 = 32
  const int h    = (blockIdx.x >> 5) & 15;
  const int b    = blockIdx.x >> 9;
  const int q0w  = qblk * 64 + wid * 16;

  const size_t headoff = ((size_t)b * H_ + h) * S_ * HD_;
  const _Float16* Qg = Qh + headoff;
  const _Float16* Kg = Kh + headoff;
  const _Float16* Vg = Vh + headoff;

  // cooperative async stage of one 32-key K/V block into buffer `buf`
  const int lin0 = (int)threadIdx.x * 8;
  const int lr0 = lin0 >> 6, lc0 = lin0 & 63;          // chunk 0
  const int lin1 = 1024 + lin0;
  const int lr1 = lin1 >> 6, lc1 = lin1 & 63;          // chunk 1

#define STAGE_KV(kb, buf)                                                      \
  do {                                                                         \
    const int kbase_ = (kb) * 32;                                              \
    async_copy_b128((unsigned)(uintptr_t)&Ks[buf][lr0][lc0],                   \
                    Kg + (size_t)(kbase_ + lr0) * HD_ + lc0);                  \
    async_copy_b128((unsigned)(uintptr_t)&Ks[buf][lr1][lc1],                   \
                    Kg + (size_t)(kbase_ + lr1) * HD_ + lc1);                  \
    async_copy_b128((unsigned)(uintptr_t)&Vs[buf][lr0][lc0],                   \
                    Vg + (size_t)(kbase_ + lr0) * HD_ + lc0);                  \
    async_copy_b128((unsigned)(uintptr_t)&Vs[buf][lr1][lc1],                   \
                    Vg + (size_t)(kbase_ + lr1) * HD_ + lc1);                  \
  } while (0)

  // Q fragments (A layout) loaded straight from global [q][hd] row-major
  v16h qf[2];
#pragma unroll
  for (int c = 0; c < 2; ++c) {
    const _Float16* base = Qg + (size_t)(q0w + nl) * HD_ + c * 32;
#pragma unroll
    for (int v = 0; v < 8; ++v) {
      const int k = (hi << 4) + (v << 1);
      qf[c][2 * v]     = base[k];
      qf[c][2 * v + 1] = base[k + 1];
    }
  }

  v8f o[4] = {};
  float rowmax[8], rowsum[8];
#pragma unroll
  for (int r = 0; r < 8; ++r) { rowmax[r] = -1e30f; rowsum[r] = 0.f; }

  STAGE_KV(0, 0);

  for (int kb = 0; kb < S_ / 32; ++kb) {
    const int kbase = kb * 32;
    const int cur = kb & 1;
    wait_async0();      // our async LDS writes for `cur` are done
    __syncthreads();    // everyone's are done; prev iteration compute finished
    if (kb + 1 < S_ / 32) STAGE_KV(kb + 1, cur ^ 1);  // overlap with compute

    // scores: two 16x16 tiles over 32 keys, contraction over HD=64 (2 frags)
    v8f s0 = {}, s1 = {};
#pragma unroll
    for (int c = 0; c < 2; ++c) {
      const v16h bk0 = load_B_fragT(&Ks[cur][0][0], 0,  c * 32, 72);
      const v16h bk1 = load_B_fragT(&Ks[cur][0][0], 16, c * 32, 72);
      s0 = wmma_f16(qf[c], bk0, s0);
      s1 = wmma_f16(qf[c], bk1, s1);
    }

    // online softmax update (row = r + 8*hi spans 16 lanes; reduce width 16)
#pragma unroll
    for (int r = 0; r < 8; ++r) {
      const int qrow = q0w + r + hi * 8;
      float x0 = s0[r] * 0.125f;               // 1/sqrt(64)
      float x1 = s1[r] * 0.125f;
      if (kbase + nl      > qrow) x0 = -1.0e-4f;   // faithful: -0.0001, not -inf
      if (kbase + 16 + nl > qrow) x1 = -1.0e-4f;
      float mx = fmaxf(x0, x1);
#pragma unroll
      for (int d = 8; d >= 1; d >>= 1) mx = fmaxf(mx, __shfl_xor(mx, d, 16));
      const float nm   = fmaxf(rowmax[r], mx);
      const float corr = __expf(rowmax[r] - nm);
      rowmax[r] = nm;
      const float p0 = __expf(x0 - nm);
      const float p1 = __expf(x1 - nm);
      rowsum[r] = rowsum[r] * corr + p0 + p1;   // lane-partial; reduced at end
#pragma unroll
      for (int t = 0; t < 4; ++t) o[t][r] *= corr;
      Ps[wid][r + hi * 8][nl]      = (_Float16)p0;
      Ps[wid][r + hi * 8][16 + nl] = (_Float16)p1;
    }
    wave_lds_sync();

    // P(16x32) x V(32x64): one A frag, four 16-wide hd tiles
    const v16h pf = load_A_frag(&Ps[wid][0][0], 0, 0, 40);
#pragma unroll
    for (int t = 0; t < 4; ++t) {
      const v16h bv = load_B_frag(&Vs[cur][0][0], 0, t * 16, 72);
      o[t] = wmma_f16(pf, bv, o[t]);
    }
  }
#undef STAGE_KV

  // finalize: reduce lane-partial row sums, normalize, write merged-head f16
#pragma unroll
  for (int r = 0; r < 8; ++r) {
    float s = rowsum[r];
#pragma unroll
    for (int d = 8; d >= 1; d >>= 1) s += __shfl_xor(s, d, 16);
    const float inv = 1.0f / s;
    const int q = q0w + r + hi * 8;
    _Float16* dst = AO + ((size_t)(b * S_ + q)) * D_ + h * HD_;
#pragma unroll
    for (int t = 0; t < 4; ++t)
      dst[t * 16 + nl] = (_Float16)(o[t][r] * inv);
  }
}

extern "C" void kernel_launch(void* const* d_in, const int* in_sizes, int n_in,
                              void* d_out, int out_size, void* d_ws, size_t ws_size,
                              hipStream_t stream) {
  (void)in_sizes; (void)n_in; (void)out_size; (void)ws_size;
  const float* X    = (const float*)d_in[0];
  const float* Wqkv = (const float*)d_in[1];
  const float* bqkv = (const float*)d_in[2];
  const float* Wd   = (const float*)d_in[3];
  const float* bd   = (const float*)d_in[4];
  // d_in[5] = causal mask: recomputed analytically in-kernel.

  constexpr size_t HSZ = (size_t)B_ * H_ * S_ * HD_;  // 4M halves per tensor
  _Float16* Qh = (_Float16*)d_ws;
  _Float16* Kh = Qh + HSZ;
  _Float16* Vh = Kh + HSZ;
  _Float16* AO = Vh + HSZ;                            // [B*S, D] f16

  const dim3 blk(128);
  // 1) QKV projection: [4096 x 1024] x [1024 x 3072] + bias -> head-split f16
  wmma_gemm_kernel<0><<<dim3((3 * D_) / 64, (B_ * S_) / 64), blk, 0, stream>>>(
      X, nullptr, Wqkv, bqkv, nullptr, Qh, Kh, Vh, B_ * S_, 3 * D_, D_);
  // 2) attention
  attention_kernel<<<dim3(B_ * H_ * (S_ / 64)), blk, 0, stream>>>(Qh, Kh, Vh, AO);
  // 3) output projection: [4096 x 1024] x [1024 x 1024] + bias -> f32 out
  wmma_gemm_kernel<1><<<dim3(D_ / 64, (B_ * S_) / 64), blk, 0, stream>>>(
      nullptr, AO, Wd, bd, (float*)d_out, nullptr, nullptr, nullptr,
      B_ * S_, D_, D_);
}
